// AttenQueryFusion_60172491817904
// MI455X (gfx1250) — matrix operand order repaired and checked by reading, not compiled, not run on hardware
//
#include <hip/hip_runtime.h>
#include <hip/hip_bf16.h>
#include <math.h>

typedef __attribute__((ext_vector_type(16))) __bf16 v16bf;
typedef __attribute__((ext_vector_type(8)))  __bf16 v8bf;
typedef __attribute__((ext_vector_type(8)))  float  v8f;

#define NKEYS   1024
#define DIM     256
#define THREADS 256
#define WAVES   8                       // THREADS / 32
#define KEYS_PER_WAVE (NKEYS / WAVES)   // 128
#define TILES_PER_WAVE (KEYS_PER_WAVE / 16) // 8
#define MGROUPS 4                       // phase-B key groups
#define MPERG   (NKEYS / MGROUPS)       // 256 keys per group

__global__ __launch_bounds__(THREADS)
void attn_query0_kernel(const float* __restrict__ x, float* __restrict__ out)
{
    // query pre-packed as bf16 hi / residual-lo (16B-aligned for ds_load_b128)
    __shared__ __align__(16) __bf16 qhb[DIM];
    __shared__ __align__(16) __bf16 qlb[DIM];
    __shared__ float sc[NKEYS];         // scores, then exp(scores - max)
    __shared__ float red[THREADS];
    __shared__ float red2[MGROUPS * DIM]; // phase-B cross-group partials

    const int t = threadIdx.x;
    const int b = blockIdx.x;
    const float* __restrict__ xb = x + (size_t)b * NKEYS * DIM;

    // ---- load q = x[b,0,:], split into bf16 hi + bf16 residual, pack in LDS ----
    {
        float qv = xb[t];
        __bf16 h = (__bf16)qv;
        qhb[t] = h;
        qlb[t] = (__bf16)(qv - (float)h);
    }
    __syncthreads();

    const int wave = t >> 5;
    const int lane = t & 31;
    const int half = lane >> 4;   // 0: lanes 0-15 (K 0-15), 1: lanes 16-31 (K 16-31)
    const int col  = lane & 15;   // B-matrix column == key within tile

    // ---- Phase A: s[m] = q . k_m / 16 via bf16 hi/lo split WMMA (~fp32 exact) ----
    for (int tile = 0; tile < TILES_PER_WAVE; ++tile) {
        const int keybase = wave * KEYS_PER_WAVE + tile * 16;
        // this lane supplies B[K = half*16 + e, col]: 16 consecutive d per chunk
        const float* krow = xb + (size_t)(keybase + col) * DIM + half * 16;

        v8f acc = {};
        #pragma unroll
        for (int c = 0; c < 8; ++c) {           // 8 chunks of K=32 cover D=256
            const float4 f0 = *(const float4*)(krow + c * 32 + 0);
            const float4 f1 = *(const float4*)(krow + c * 32 + 4);
            const float4 f2 = *(const float4*)(krow + c * 32 + 8);
            const float4 f3 = *(const float4*)(krow + c * 32 + 12);
            const float f[16] = { f0.x, f0.y, f0.z, f0.w,
                                  f1.x, f1.y, f1.z, f1.w,
                                  f2.x, f2.y, f2.z, f2.w,
                                  f3.x, f3.y, f3.z, f3.w };
            v16bf bhi, blo;
            #pragma unroll
            for (int i = 0; i < 16; ++i) {
                __bf16 h = (__bf16)f[i];
                bhi[i] = h;
                blo[i] = (__bf16)(f[i] - (float)h);
            }
            // A fragment (replicated q), contiguous k order (not the ISA table order):
            // lanes<16: elems 0-15 = K c*32+0..15 ; lanes>=16: K c*32+16..31 (matches B)
            const __bf16* qp = qhb + c * 32 + half * 16;
            const __bf16* rp = qlb + c * 32 + half * 16;
            v8bf h0 = *(const v8bf*)(qp);
            v8bf h1 = *(const v8bf*)(qp + 8);
            v8bf l0 = *(const v8bf*)(rp);
            v8bf l1 = *(const v8bf*)(rp + 8);
            v16bf ahi = __builtin_shufflevector(h0, h1, 0,1,2,3,4,5,6,7,
                                                         8,9,10,11,12,13,14,15);
            v16bf alo = __builtin_shufflevector(l0, l1, 0,1,2,3,4,5,6,7,
                                                         8,9,10,11,12,13,14,15);
            acc = __builtin_amdgcn_wmma_f32_16x16x32_bf16(false, ahi, false, bhi,
                                                          (short)0, acc, false, false);
            acc = __builtin_amdgcn_wmma_f32_16x16x32_bf16(false, ahi, false, blo,
                                                          (short)0, acc, false, false);
            acc = __builtin_amdgcn_wmma_f32_16x16x32_bf16(false, alo, false, bhi,
                                                          (short)0, acc, false, false);
        }
        // C layout: VGPR0 lanes 0-15 = (M=0, N=lane); all A rows identical
        if (lane < 16) sc[keybase + lane] = acc[0] * 0.0625f;  // / sqrt(256)
    }
    __syncthreads();

    // ---- block softmax over the 1024 scores ----
    float lmax = -INFINITY;
    #pragma unroll
    for (int r = 0; r < NKEYS / THREADS; ++r)
        lmax = fmaxf(lmax, sc[t + r * THREADS]);
    red[t] = lmax;
    __syncthreads();
    for (int off = THREADS / 2; off > 0; off >>= 1) {
        if (t < off) red[t] = fmaxf(red[t], red[t + off]);
        __syncthreads();
    }
    const float mx = red[0];
    __syncthreads();

    float lsum = 0.f;
    #pragma unroll
    for (int r = 0; r < NKEYS / THREADS; ++r) {
        float e = expf(sc[t + r * THREADS] - mx);
        sc[t + r * THREADS] = e;
        lsum += e;
    }
    red[t] = lsum;
    __syncthreads();
    for (int off = THREADS / 2; off > 0; off >>= 1) {
        if (t < off) red[t] += red[t + off];
        __syncthreads();
    }
    const float inv = 1.0f / red[0];
    __syncthreads();

    // ---- Phase B: out[d] = inv * sum_m p[m] * x[b,m,d]  (exact fp32 VALU) ----
    // thread -> (key-group g, d-quad tq); one float4 load per key per thread
    {
        const int g  = t >> 6;          // 0..3
        const int tq = t & 63;          // d = tq*4 .. tq*4+3
        const int mbase = g * MPERG;
        const float* __restrict__ xg = xb + (size_t)mbase * DIM + tq * 4;

        float4 a4 = make_float4(0.f, 0.f, 0.f, 0.f);
        for (int mm = 0; mm < MPERG; mm += 8) {
            if (mm + 64 < MPERG)
                __builtin_prefetch(xg + (size_t)(mm + 64) * DIM, 0, 0);
            #pragma unroll
            for (int j = 0; j < 8; ++j) {
                const float  p = sc[mbase + mm + j];
                const float4 v = *(const float4*)(xg + (size_t)(mm + j) * DIM);
                a4.x = fmaf(p, v.x, a4.x);
                a4.y = fmaf(p, v.y, a4.y);
                a4.z = fmaf(p, v.z, a4.z);
                a4.w = fmaf(p, v.w, a4.w);
            }
        }
        float* rp = red2 + g * DIM + tq * 4;
        rp[0] = a4.x; rp[1] = a4.y; rp[2] = a4.z; rp[3] = a4.w;
    }
    __syncthreads();

    const float o = (red2[t] + red2[DIM + t]) + (red2[2 * DIM + t] + red2[3 * DIM + t]);
    out[(size_t)b * DIM + t] = o * inv;
}

extern "C" void kernel_launch(void* const* d_in, const int* in_sizes, int n_in,
                              void* d_out, int out_size, void* d_ws, size_t ws_size,
                              hipStream_t stream)
{
    (void)n_in; (void)d_ws; (void)ws_size; (void)out_size;
    const float* x = (const float*)d_in[0];
    float* out = (float*)d_out;
    const int B = in_sizes[0] / (NKEYS * DIM);   // 256 for the reference shapes
    attn_query0_kernel<<<B, THREADS, 0, stream>>>(x, out);
}
